// LlamaGQA_57440892617053
// MI455X (gfx1250) — matrix operand order repaired and checked by reading, not compiled, not run on hardware
//
#include <hip/hip_runtime.h>
#include <hip/hip_bf16.h>

// ---------------------------------------------------------------------------
// Llama GQA block for gfx1250 (MI455X): bf16 WMMA + TDM (tensor_load_to_lds)
// double-buffered GEMM staging, flash attention with WMMA.
// ---------------------------------------------------------------------------

typedef __bf16 bf16_t;
typedef __bf16 v4bf  __attribute__((ext_vector_type(4)));
typedef __bf16 v8bf  __attribute__((ext_vector_type(8)));
typedef __bf16 v16bf __attribute__((ext_vector_type(16)));
typedef float  v8f   __attribute__((ext_vector_type(8)));
typedef unsigned int u32x4 __attribute__((ext_vector_type(4)));
typedef int    i32x4 __attribute__((ext_vector_type(4)));
typedef int    i32x8 __attribute__((ext_vector_type(8)));

union Frag { v16bf v; v8bf h[2]; };

// Problem constants
constexpr int DIMC     = 4096;
constexpr int NHEADS   = 32;
constexpr int NKV      = 8;
constexpr int HDIM     = 128;
constexpr int BATCH    = 2;
constexpr int SLEN     = 2048;
constexpr int TOK      = BATCH * SLEN;      // 4096 tokens
constexpr int KVDIM    = NKV * HDIM;        // 1024
constexpr float ASCALE = 0.08838834764831845f; // 1/sqrt(128)

// Workspace layout (bytes)
constexpr size_t OFF_XB = 0;                              // x bf16      32MB
constexpr size_t OFF_QB = 33554432ull;                    // q           32MB
constexpr size_t OFF_KB = OFF_QB + 33554432ull;           // k            8MB
constexpr size_t OFF_VB = OFF_KB + 8388608ull;            // v            8MB
constexpr size_t OFF_AB = OFF_VB + 8388608ull;            // attn out    32MB
constexpr size_t OFF_WQ = OFF_AB + 33554432ull;           // wq bf16     32MB
constexpr size_t OFF_WK = OFF_WQ + 33554432ull;           // wk bf16      8MB
constexpr size_t OFF_WV = OFF_WK + 8388608ull;            // wv bf16      8MB
constexpr size_t OFF_WO = OFF_WV + 8388608ull;            // wo bf16     32MB

// ---------------------------------------------------------------------------
// WMMA helpers (CDNA5 16x16x32 bf16, f32 accumulate)
// ---------------------------------------------------------------------------
__device__ __forceinline__ v8f wmma_bf16(v16bf a, v16bf b, v8f c) {
  return __builtin_amdgcn_wmma_f32_16x16x32_bf16(false, a, false, b,
                                                 (short)0, c, false, false);
}

// A-matrix 16x32 fragment: lane<16 holds M=lane, K={0..7,16..23};
// lane>=16 holds M=lane-16, K={8..15,24..31}  (ISA 7.12.2)
__device__ __forceinline__ v16bf ld_fragA(const bf16_t* t, int row0, int col0,
                                          int stride, int lane) {
  const bf16_t* p = t + (size_t)(row0 + (lane & 15)) * stride + col0 +
                    ((lane & 16) ? 8 : 0);
  Frag f;
  f.h[0] = *(const v8bf*)p;
  f.h[1] = *(const v8bf*)(p + 16);
  return f.v;
}

// B-matrix 32x16 fragment loaded from B^T (row-major [N][K]):
// lane<16 holds N=lane, K={0..7,16..23}; lane>=16 holds N=lane-16, K={8..15,24..31}
__device__ __forceinline__ v16bf ld_fragB(const bf16_t* t, int n0, int col0,
                                          int stride, int lane) {
  const bf16_t* p = t + (size_t)(n0 + (lane & 15)) * stride + col0 +
                    ((lane & 16) ? 8 : 0);
  Frag f;
  f.h[0] = *(const v8bf*)p;
  f.h[1] = *(const v8bf*)(p + 16);
  return f.v;
}

// ---------------------------------------------------------------------------
// TDM: issue a 2-D bf16 tile load (rows x cols) global -> LDS.
// D# per CDNA5 ISA ch. 8: group0 = {count, lds_addr, global_addr, type=2},
// group1 = {data_size=2B, tensor dims (large: no OOB clip), tile dims,
// tensor_dim0_stride}, groups 2/3 zero (2-D tensor).
// One wave issues; completion tracked in-order on TENSORcnt.
// Toolchain here is the 6-arg clang-23 builtin:
//   (u32x4 g0, i32x8 g1, i32x4 g2, i32x4 g3, i32x8 g4, i32 cpol)
// ---------------------------------------------------------------------------
__device__ __forceinline__ void tdm_load_tile_2d(const bf16_t* gsrc,
                                                 unsigned int lds_off,
                                                 int cols, int rows,
                                                 long long strideElems) {
  const unsigned long long ga = (unsigned long long)(uintptr_t)gsrc;
  const unsigned int td0 = 0x7FFFFFFFu;   // tensor_dim0 (huge -> no clipping)
  const unsigned int td1 = 0x7FFFFFFFu;   // tensor_dim1
  const unsigned long long st0 = (unsigned long long)strideElems;

  u32x4 g0;
  g0[0] = 1u;                                         // count=1, user D#
  g0[1] = lds_off;                                    // lds_addr
  g0[2] = (unsigned int)(ga & 0xFFFFFFFFull);         // global_addr[31:0]
  g0[3] = (unsigned int)((ga >> 32) & 0x1FFFFFFull)   // global_addr[56:32]
          | (2u << 30);                               // type=2 ("image")

  i32x8 g1;
  g1[0] = (int)(1u << 16);                            // data_size=1 (2 bytes)
  g1[1] = (int)((td0 & 0xFFFFu) << 16);               // tensor_dim0[15:0]
  g1[2] = (int)(((td0 >> 16) & 0xFFFFu) |
                ((td1 & 0xFFFFu) << 16));             // td0 hi | td1 lo
  g1[3] = (int)(((td1 >> 16) & 0xFFFFu) |
                (((unsigned)cols & 0xFFFFu) << 16));  // td1 hi | tile_dim0
  g1[4] = (int)((unsigned)rows & 0xFFFFu);            // tile_dim1 (tile_dim2=0)
  g1[5] = (int)(unsigned int)(st0 & 0xFFFFFFFFull);   // dim0_stride[31:0]
  g1[6] = (int)(unsigned int)((st0 >> 32) & 0xFFFFull); // dim0_stride[47:32]
  g1[7] = 0;

  i32x4 gz4 = {0, 0, 0, 0};
  i32x8 gz8 = {0, 0, 0, 0, 0, 0, 0, 0};
  __builtin_amdgcn_tensor_load_to_lds(g0, g1, gz4, gz4, gz8, 0);
}

// ---------------------------------------------------------------------------
// fp32 -> bf16 conversion (vectorized)
// ---------------------------------------------------------------------------
__global__ void gqa_cvt_bf16(const float* __restrict__ s,
                             bf16_t* __restrict__ d, long long n4) {
  long long i  = (long long)blockIdx.x * blockDim.x + threadIdx.x;
  long long st = (long long)gridDim.x * blockDim.x;
  for (; i < n4; i += st) {
    float4 f = ((const float4*)s)[i];
    v4bf o;
    o[0] = (bf16_t)f.x; o[1] = (bf16_t)f.y;
    o[2] = (bf16_t)f.z; o[3] = (bf16_t)f.w;
    ((v4bf*)d)[i] = o;
  }
}

// ---------------------------------------------------------------------------
// Tiled GEMM: C[M,N] = A[M,K] @ Bw[N,K]^T  (Bw row-major = torch Linear W)
// Block 128x128, BK=64, 256 threads (8 waves: 4x2), wave tile 32x64.
// HBM -> LDS staging via TDM, double-buffered; wave0 drives the DMA.
// ---------------------------------------------------------------------------
constexpr int GBM = 128, GBN = 128, GBK = 64, GST = GBK;  // unpadded TDM rows

template <typename OutT>
__global__ __launch_bounds__(256)
void gqa_gemm_bf16(const bf16_t* __restrict__ A, const bf16_t* __restrict__ Bw,
                   OutT* __restrict__ C, int M, int N, int K) {
  __shared__ __align__(16) bf16_t As[2][GBM * GST];   // 2 x 16KB
  __shared__ __align__(16) bf16_t Bs[2][GBN * GST];   // 2 x 16KB
  const int tid  = threadIdx.x;
  const int lane = tid & 31;
  const int wave = tid >> 5;
  const int wm   = wave >> 1;  // 0..3 : 32 rows each
  const int wn   = wave & 1;   // 0..1 : 64 cols each
  const int m0   = blockIdx.y * GBM;
  const int n0   = blockIdx.x * GBN;

  const bf16_t* arow = A + (size_t)m0 * K;
  const bf16_t* brow = Bw + (size_t)n0 * K;

  v8f acc[2][4] = {};

  const int nk = K / GBK;
  // Prologue: stage tile 0 into buffer 0.
  if (wave == 0) {
    tdm_load_tile_2d(arow, (unsigned int)(uintptr_t)&As[0][0], GBK, GBM, K);
    tdm_load_tile_2d(brow, (unsigned int)(uintptr_t)&Bs[0][0], GBK, GBN, K);
  }

  for (int t = 0; t < nk; ++t) {
    const int cur = t & 1;
    if (wave == 0) {
      if (t + 1 < nk) {
        // Prefetch next tile into the other buffer (reads of it finished at
        // the end-of-iteration barrier of t-1).
        tdm_load_tile_2d(arow + (size_t)(t + 1) * GBK,
                         (unsigned int)(uintptr_t)&As[cur ^ 1][0], GBK, GBM, K);
        tdm_load_tile_2d(brow + (size_t)(t + 1) * GBK,
                         (unsigned int)(uintptr_t)&Bs[cur ^ 1][0], GBK, GBN, K);
        // TENSORcnt is in-order: <=2 outstanding means tile t has landed.
        __builtin_amdgcn_s_wait_tensorcnt(2);
      } else {
        __builtin_amdgcn_s_wait_tensorcnt(0);
      }
    }
    __syncthreads();

    #pragma unroll
    for (int kk = 0; kk < GBK; kk += 32) {
      v16bf a0 = ld_fragA(As[cur], wm * 32,      kk, GST, lane);
      v16bf a1 = ld_fragA(As[cur], wm * 32 + 16, kk, GST, lane);
      #pragma unroll
      for (int j = 0; j < 4; ++j) {
        v16bf b = ld_fragB(Bs[cur], wn * 64 + j * 16, kk, GST, lane);
        acc[0][j] = wmma_bf16(a0, b, acc[0][j]);
        acc[1][j] = wmma_bf16(a1, b, acc[1][j]);
      }
    }
    __syncthreads();
  }

  // Epilogue: C layout — VGPR e: lanes0-15 M=e, lanes16-31 M=e+8; N=lane&15.
  const int rofs = (lane & 16) ? 8 : 0;
  const int cn   = n0 + wn * 64 + (lane & 15);
  #pragma unroll
  for (int i = 0; i < 2; ++i)
    #pragma unroll
    for (int j = 0; j < 4; ++j)
      #pragma unroll
      for (int e = 0; e < 8; ++e) {
        int r = m0 + wm * 32 + i * 16 + e + rofs;
        C[(size_t)r * N + cn + j * 16] = (OutT)acc[i][j][e];
      }
}

// ---------------------------------------------------------------------------
// Flash attention (no causal mask, GQA). One block = (b, head, 64-query tile),
// 4 waves of 32. Key tile = 64, online softmax in f32, WMMA for S and P@V.
// ---------------------------------------------------------------------------
constexpr int AQT = 64, AKT = 64;
constexpr int SQ = HDIM + 8;   // 136
constexpr int SK = HDIM + 8;   // 136
constexpr int SV = AKT + 8;    // 72
constexpr int SP = AKT + 8;    // 72

__global__ __launch_bounds__(128)
void gqa_attention(const bf16_t* __restrict__ Q, const bf16_t* __restrict__ Kb,
                   const bf16_t* __restrict__ Vb, bf16_t* __restrict__ Ob) {
  __shared__ __align__(16) bf16_t Qs[AQT * SQ];
  __shared__ __align__(16) bf16_t Ks[AKT * SK];
  __shared__ __align__(16) bf16_t Vt[HDIM * SV];   // V^T: [d][kcol]
  __shared__ __align__(16) bf16_t Ps[AQT * SP];

  const int tid  = threadIdx.x;
  const int lane = tid & 31;
  const int w    = tid >> 5;                  // wave 0..3 -> 16 q-rows each
  const int nqt  = SLEN / AQT;                // 32
  const int qt   = blockIdx.x % nqt;
  const int h    = (blockIdx.x / nqt) % NHEADS;
  const int b    = blockIdx.x / (nqt * NHEADS);
  const int hk   = h / (NHEADS / NKV);
  const size_t tok0 = (size_t)b * SLEN + (size_t)qt * AQT;

  // Stage 64x128 Q tile once.
  #pragma unroll
  for (int it = 0; it < 8; ++it) {
    int idx = it * 128 + tid;   // 0..1023
    int row = idx >> 4;
    int ch  = (idx & 15) * 8;
    *(uint4*)&Qs[row * SQ + ch] =
        *(const uint4*)&Q[(tok0 + row) * (NHEADS * HDIM) + h * HDIM + ch];
  }

  float mrow[8], lrow[8];
  #pragma unroll
  for (int i = 0; i < 8; ++i) { mrow[i] = -1e30f; lrow[i] = 0.f; }
  v8f Ot[8] = {};
  const int rofs = (lane & 16) ? 8 : 0;

  for (int kb0 = 0; kb0 < SLEN; kb0 += AKT) {
    __syncthreads();  // previous iteration fully consumed K/V/P tiles
    const size_t ktok0 = (size_t)b * SLEN + kb0;
    // Stage K tile [kcol][d] and V^T tile [d][kcol].
    #pragma unroll
    for (int it = 0; it < 8; ++it) {
      int idx = it * 128 + tid;
      int row = idx >> 4;           // key col 0..63
      int ch  = (idx & 15) * 8;     // d chunk
      const size_t srcOff = (ktok0 + row) * KVDIM + hk * HDIM + ch;
      *(uint4*)&Ks[row * SK + ch] = *(const uint4*)&Kb[srcOff];
      v8bf vv = *(const v8bf*)&Vb[srcOff];
      #pragma unroll
      for (int e = 0; e < 8; ++e) Vt[(ch + e) * SV + row] = vv[e];
    }
    __syncthreads();

    // Scores: 16 q-rows x 64 key-cols, accumulate over d=128 in 4 WMMA steps.
    v8f sc[4] = {};
    #pragma unroll
    for (int kk = 0; kk < HDIM; kk += 32) {
      v16bf a = ld_fragA(Qs, w * 16, kk, SQ, lane);
      #pragma unroll
      for (int j = 0; j < 4; ++j) {
        v16bf bb = ld_fragB(Ks, j * 16, kk, SK, lane);
        sc[j] = wmma_bf16(a, bb, sc[j]);
      }
    }

    // Online softmax (row reductions across 16-lane halves via shfl_xor).
    #pragma unroll
    for (int i = 0; i < 8; ++i) {
      float mx = fmaxf(fmaxf(sc[0][i], sc[1][i]), fmaxf(sc[2][i], sc[3][i]));
      mx = fmaxf(mx, __shfl_xor(mx, 1));
      mx = fmaxf(mx, __shfl_xor(mx, 2));
      mx = fmaxf(mx, __shfl_xor(mx, 4));
      mx = fmaxf(mx, __shfl_xor(mx, 8));
      float mnew = fmaxf(mrow[i], mx);
      float corr = __expf(ASCALE * (mrow[i] - mnew));
      mrow[i] = mnew;
      float rs = 0.f;
      #pragma unroll
      for (int j = 0; j < 4; ++j) {
        float p = __expf(ASCALE * (sc[j][i] - mnew));
        sc[j][i] = p;
        rs += p;
      }
      rs += __shfl_xor(rs, 1);
      rs += __shfl_xor(rs, 2);
      rs += __shfl_xor(rs, 4);
      rs += __shfl_xor(rs, 8);
      lrow[i] = lrow[i] * corr + rs;
      #pragma unroll
      for (int t = 0; t < 8; ++t) Ot[t][i] *= corr;
    }

    // P -> LDS as bf16 (per-wave private rows; DS ops are in-order per wave).
    #pragma unroll
    for (int j = 0; j < 4; ++j)
      #pragma unroll
      for (int i = 0; i < 8; ++i)
        Ps[(w * 16 + i + rofs) * SP + j * 16 + (lane & 15)] = (bf16_t)sc[j][i];

    // O += P @ V : accumulate over 64 key-cols in 2 WMMA steps per d-tile.
    #pragma unroll
    for (int kk = 0; kk < AKT; kk += 32) {
      v16bf a = ld_fragA(Ps, w * 16, kk, SP, lane);
      #pragma unroll
      for (int t = 0; t < 8; ++t) {
        v16bf bb = ld_fragB(Vt, t * 16, kk, SV, lane);
        Ot[t] = wmma_bf16(a, bb, Ot[t]);
      }
    }
  }

  // Normalize and write attention output [tok, h*128+d] as bf16.
  #pragma unroll
  for (int i = 0; i < 8; ++i) lrow[i] = 1.0f / lrow[i];
  #pragma unroll
  for (int t = 0; t < 8; ++t)
    #pragma unroll
    for (int i = 0; i < 8; ++i) {
      size_t tok = tok0 + w * 16 + i + rofs;
      int col = h * HDIM + t * 16 + (lane & 15);
      Ob[tok * (NHEADS * HDIM) + col] = (bf16_t)(Ot[t][i] * lrow[i]);
    }
}

// ---------------------------------------------------------------------------
// Host-side launch
// ---------------------------------------------------------------------------
extern "C" void kernel_launch(void* const* d_in, const int* in_sizes, int n_in,
                              void* d_out, int out_size, void* d_ws,
                              size_t ws_size, hipStream_t stream) {
  (void)in_sizes; (void)n_in; (void)out_size; (void)ws_size;
  const float* x  = (const float*)d_in[0];
  const float* wq = (const float*)d_in[1];
  const float* wk = (const float*)d_in[2];
  const float* wv = (const float*)d_in[3];
  const float* wo = (const float*)d_in[4];

  char* ws = (char*)d_ws;
  bf16_t* xb  = (bf16_t*)(ws + OFF_XB);
  bf16_t* qb  = (bf16_t*)(ws + OFF_QB);
  bf16_t* kb  = (bf16_t*)(ws + OFF_KB);
  bf16_t* vb  = (bf16_t*)(ws + OFF_VB);
  bf16_t* ab  = (bf16_t*)(ws + OFF_AB);
  bf16_t* wqb = (bf16_t*)(ws + OFF_WQ);
  bf16_t* wkb = (bf16_t*)(ws + OFF_WK);
  bf16_t* wvb = (bf16_t*)(ws + OFF_WV);
  bf16_t* wob = (bf16_t*)(ws + OFF_WO);

  auto cvt = [&](const float* s, bf16_t* d, long long n) {
    long long n4 = n / 4;
    long long g  = (n4 + 255) / 256;
    if (g > 4096) g = 4096;
    hipLaunchKernelGGL(gqa_cvt_bf16, dim3((unsigned)g), dim3(256), 0, stream,
                       s, d, n4);
  };
  cvt(x,  xb,  (long long)TOK * DIMC);
  cvt(wq, wqb, (long long)DIMC * DIMC);
  cvt(wk, wkb, (long long)KVDIM * DIMC);
  cvt(wv, wvb, (long long)KVDIM * DIMC);
  cvt(wo, wob, (long long)DIMC * DIMC);

  // Projections: q = x @ wq^T, k = x @ wk^T, v = x @ wv^T  (bf16 out)
  hipLaunchKernelGGL((gqa_gemm_bf16<bf16_t>), dim3(DIMC / GBN, TOK / GBM),
                     dim3(256), 0, stream, xb, wqb, qb, TOK, DIMC, DIMC);
  hipLaunchKernelGGL((gqa_gemm_bf16<bf16_t>), dim3(KVDIM / GBN, TOK / GBM),
                     dim3(256), 0, stream, xb, wkb, kb, TOK, KVDIM, DIMC);
  hipLaunchKernelGGL((gqa_gemm_bf16<bf16_t>), dim3(KVDIM / GBN, TOK / GBM),
                     dim3(256), 0, stream, xb, wvb, vb, TOK, KVDIM, DIMC);

  // Attention: grid = B * H * (S / 64) = 2048 blocks of 128 threads.
  hipLaunchKernelGGL(gqa_attention,
                     dim3(BATCH * NHEADS * (SLEN / AQT)), dim3(128), 0, stream,
                     qb, kb, vb, ab);

  // Output projection: out = attn @ wo^T  (f32 out straight to d_out)
  hipLaunchKernelGGL((gqa_gemm_bf16<float>), dim3(DIMC / GBN, TOK / GBM),
                     dim3(256), 0, stream, ab, wob, (float*)d_out,
                     TOK, DIMC, DIMC);
}
